// Matcher_48232482734540
// MI455X (gfx1250) — compile-verified, matrix operand order checked by reading, not run on hardware
//
#include <hip/hip_runtime.h>
#include <hip/hip_bf16.h>

// ---------------------------------------------------------------------------
// Flash-attention style memory-bank matcher for MI455X (gfx1250, wave32).
//   logits = K^T Q / sqrt(Dk); P = softmax_banks; mem = V P; gate = mask . P
//   out = concat(mem, q_out * gate)  -> (1, O, 2*Dv, N)
// bf16 WMMA (v_wmma_f32_16x16x32_bf16) + f32 accum, online softmax.
// K/V tiles streamed to LDS by the Tensor Data Mover (tensor_load_to_lds),
// double-buffered, TENSORcnt-synchronized; cooperative-load fallback if the
// TDM builtin is unavailable.
// ---------------------------------------------------------------------------

typedef __attribute__((ext_vector_type(16))) __bf16 v16bf;
typedef __attribute__((ext_vector_type(8)))  float  v8f;
typedef __attribute__((ext_vector_type(4)))  unsigned int u32x4;
typedef __attribute__((ext_vector_type(8)))  int    i32x8;
typedef __attribute__((ext_vector_type(4)))  int    i32x4;

#define OBJ_N  4
#define D_KEY  128
#define D_VAL  512
#define BANK_N 7200
#define N_Q    3600
#define QT     16      // queries per workgroup (WMMA N)
#define CK     32      // banks per chunk (WMMA K for bf16)
#define NWAVE  4
#define NTHREADS 128
#define NCH    (BANK_N / CK)   // 225 chunks
#define ROWP   34      // padded LDS row stride in floats (32 data + 2 pad)

// Dynamic LDS layout (bytes)
#define SM_K_FLOATS (2 * D_KEY * ROWP)          //  8704 f = 34.8 KB (dbl buf)
#define SM_V_FLOATS (2 * D_VAL * ROWP)          // 34816 f = 139 KB (dbl buf)
#define SM_P_ELEMS  (NWAVE * QT * CK)           // bf16 P staging
#define SMEM_BYTES  ((SM_K_FLOATS + SM_V_FLOATS) * 4 + SM_P_ELEMS * 2 + QT * 4)

#if __has_builtin(__builtin_amdgcn_tensor_load_to_lds) && \
    __has_builtin(__builtin_amdgcn_s_wait_tensorcnt)
#define USE_TDM 1
#else
#define USE_TDM 0
#endif

// K-pair base for the bf16 A/B operand register layout (ISA 7.12.2):
// lanes 0-15: v0..3 -> K 0..7, v4..7 -> K 16..23 ; lanes 16-31: +8 / +24.
__device__ __forceinline__ int kpair_base(int v, int h) {
    return ((v < 4) ? 0 : 16) + h * 8 + (v & 3) * 2;
}

__device__ __forceinline__ unsigned lds_byte_off(const void* p) {
    // Flat addresses of LDS carry the byte offset in the low 32 bits.
    return (unsigned)(unsigned long long)(size_t)p;
}

#if USE_TDM
// Issue one 2D TDM tile load: tile (tile_d0 x tile_d1) of 4-byte elements from
// a row-major tensor (x contiguous, line stride `stride0`), padded in LDS by
// 2 DWORDs every 32 DWORDs -> LDS row stride ROWP floats. ISA ch.8 D# layout.
__device__ __forceinline__ void tdm_load_2d(unsigned lds_off, const void* gptr,
                                            unsigned tile_d0, unsigned tile_d1,
                                            unsigned tensor_d0, unsigned tensor_d1,
                                            unsigned stride0) {
    unsigned long long ga = (unsigned long long)(size_t)gptr;
    u32x4 g0 = (u32x4){};
    g0[0] = 1u;                                   // count=1, user mode
    g0[1] = lds_off;                              // lds_addr (bytes)
    g0[2] = (unsigned)(ga & 0xFFFFFFFFu);         // global_addr[31:0]
    g0[3] = (unsigned)((ga >> 32) & 0x01FFFFFFu)  // global_addr[56:32]
          | (2u << 30);                           // type = 2 ("image")
    i32x8 g1 = (i32x8){};
    g1[0] = (int)((2u << 16)     // data_size = 4 bytes
          |       (1u << 20)     // pad_enable
          |       (4u << 22)     // pad_interval: every 32 DWORDs
          |       (1u << 25));   // pad_amount: 2 DWORDs
    g1[1] = (int)((tensor_d0 & 0xFFFFu) << 16);                       // dim0 lo
    g1[2] = (int)((tensor_d0 >> 16) | ((tensor_d1 & 0xFFFFu) << 16)); // dim0 hi|dim1 lo
    g1[3] = (int)((tensor_d1 >> 16) | (tile_d0 << 16));               // dim1 hi|tile0
    g1[4] = (int)(tile_d1 & 0xFFFFu);                                 // tile1 (tile2=0)
    g1[5] = (int)stride0;                                             // dim0_stride lo
    g1[6] = 0;                                                        // stride hi / dim1_stride
    g1[7] = 0;
    i32x4 gz = (i32x4){};
#if __clang_major__ >= 23
    i32x8 gz8 = (i32x8){};
    __builtin_amdgcn_tensor_load_to_lds(g0, g1, gz, gz, gz8, 0);
#else
    __builtin_amdgcn_tensor_load_to_lds(g0, g1, gz, gz, 0);
#endif
}
#endif  // USE_TDM

// Stage one 32-bank chunk of K (128 x 32) and V (512 x 32) into padded LDS
// buffers laid out [row][ROWP] (row = d for K, = Dv row for V).
__device__ __forceinline__ void stage_chunk(int chunk, int o, int tid, int wave,
                                            const float* __restrict__ keys,
                                            const float* __restrict__ values,
                                            float* Kdst, float* Vdst) {
    const int cb = chunk * CK;
    const size_t kobase = (size_t)o * D_KEY * BANK_N;
    const size_t vobase = (size_t)o * D_VAL * BANK_N;
#if USE_TDM
    if (wave == 0) {   // TDM is per-wave, EXEC ignored; one issuing wave
        tdm_load_2d(lds_byte_off(Kdst), keys + kobase + cb,
                    CK, D_KEY, BANK_N, D_KEY, BANK_N);
        tdm_load_2d(lds_byte_off(Vdst), values + vobase + cb,
                    CK, D_VAL, BANK_N, D_VAL, BANK_N);
    }
    (void)tid;
#else
    #pragma unroll 4
    for (int i = tid; i < CK * D_KEY; i += NTHREADS) {
        const int b = i & (CK - 1), d = i >> 5;
        Kdst[d * ROWP + b] = keys[kobase + (size_t)d * BANK_N + cb + b];
    }
    #pragma unroll 8
    for (int i = tid; i < CK * D_VAL; i += NTHREADS) {
        const int b = i & (CK - 1), r = i >> 5;
        Vdst[r * ROWP + b] = values[vobase + (size_t)r * BANK_N + cb + b];
    }
#endif
}

__global__ __launch_bounds__(NTHREADS)
void Matcher_48232482734540_kernel(const float* __restrict__ keys,
                                   const float* __restrict__ values,
                                   const float* __restrict__ masks,
                                   const float* __restrict__ q_in,
                                   const float* __restrict__ q_out,
                                   float* __restrict__ out) {
    extern __shared__ __align__(16) char smem_raw[];
    float*  ldsK       = (float*)smem_raw;                   // [2][D_KEY*ROWP]
    float*  ldsV       = ldsK + SM_K_FLOATS;                 // [2][D_VAL*ROWP]
    __bf16* ldsPbase   = (__bf16*)(ldsV + SM_V_FLOATS);      // [NWAVE][QT*CK]
    float*  ldsMaskOut = (float*)(ldsPbase + SM_P_ELEMS);    // [QT]

    const int tid  = threadIdx.x;
    const int lane = tid & 31;
    const int wave = tid >> 5;
    const int q    = lane & 15;   // query col (B/C col, A row)
    const int h    = lane >> 4;   // lane half

    const int o     = blockIdx.y;
    const int nbase = blockIdx.x * QT;

    // 1/sqrt(Dk) * log2(e): softmax in base-2 domain (v_exp_f32 is exp2).
    const float qscale = 0.08838834764831845f * 1.4426950408889634f;

    // ---- Q tile in registers, WMMA-B layout, 4 slices of K=32 -------------
    v16bf Qb[4];
    #pragma unroll
    for (int kk = 0; kk < 4; ++kk) {
        #pragma unroll
        for (int v = 0; v < 8; ++v) {
            const int d = kk * 32 + kpair_base(v, h);
            const float f0 = q_in[(size_t)(d    ) * N_Q + nbase + q] * qscale;
            const float f1 = q_in[(size_t)(d + 1) * N_Q + nbase + q] * qscale;
            Qb[kk][2 * v]     = (__bf16)f0;
            Qb[kk][2 * v + 1] = (__bf16)f1;
        }
    }

    v8f acc[8];
    #pragma unroll
    for (int t = 0; t < 8; ++t) acc[t] = (v8f){};
    float m_run = -1e30f, l_run = 0.0f, maskacc = 0.0f;

    const float* mrow_base = masks + (size_t)o * BANK_N;

    // ---- Prologue: stage chunk 0 into buffer 0 ----------------------------
    stage_chunk(0, o, tid, wave, keys, values, ldsK, ldsV);
#if USE_TDM
    if (wave == 0) __builtin_amdgcn_s_wait_tensorcnt((short)0);
#endif
    __syncthreads();

    for (int i = 0; i < NCH; ++i) {
        const int par = i & 1;
        const int cb  = i * CK;
        const float* Kbuf = ldsK + par * (D_KEY * ROWP);
        const float* Vbuf = ldsV + par * (D_VAL * ROWP);

        // Kick off DMA for the next chunk into the other buffer (overlaps
        // with this chunk's WMMA work; buffer was released by last barrier).
        if (i + 1 < NCH) {
            stage_chunk(i + 1, o, tid, wave, keys, values,
                        ldsK + (1 - par) * (D_KEY * ROWP),
                        ldsV + (1 - par) * (D_VAL * ROWP));
        }
        if (wave == 0 && cb + CK < BANK_N)
            __builtin_prefetch(&mrow_base[cb + CK + lane], 0, 1);

        // ---- S = K^T Q: two 16-bank subtiles, Dk=128 -> 4 WMMAs each -----
        v8f s[2];
        #pragma unroll
        for (int sub = 0; sub < 2; ++sub) {
            v8f c = (v8f){};
            #pragma unroll
            for (int kk = 0; kk < 4; ++kk) {
                v16bf A;
                #pragma unroll
                for (int v = 0; v < 8; ++v) {
                    const int d = kk * 32 + kpair_base(v, h);
                    const float f0 = Kbuf[(size_t)(d    ) * ROWP + sub * 16 + q];
                    const float f1 = Kbuf[(size_t)(d + 1) * ROWP + sub * 16 + q];
                    A[2 * v]     = (__bf16)f0;
                    A[2 * v + 1] = (__bf16)f1;
                }
                c = __builtin_amdgcn_wmma_f32_16x16x32_bf16(
                        false, A, false, Qb[kk], (short)0, c, false, false);
            }
            s[sub] = c;
        }

        // ---- Online softmax over this chunk's 32 banks --------------------
        float cmax = -1e30f;
        #pragma unroll
        for (int v = 0; v < 8; ++v) cmax = fmaxf(cmax, fmaxf(s[0][v], s[1][v]));
        cmax = fmaxf(cmax, __shfl_xor(cmax, 16, 32));

        const float m_new = fmaxf(m_run, cmax);
        const float r     = exp2f(m_run - m_new);

        float p0[8], p1[8];
        float csum = 0.0f;
        #pragma unroll
        for (int v = 0; v < 8; ++v) {
            p0[v] = exp2f(s[0][v] - m_new);
            p1[v] = exp2f(s[1][v] - m_new);
            csum += p0[v] + p1[v];
        }
        csum += __shfl_xor(csum, 16, 32);
        l_run = l_run * r + csum;
        m_run = m_new;

        #pragma unroll
        for (int t = 0; t < 8; ++t)
            #pragma unroll
            for (int j = 0; j < 8; ++j) acc[t][j] *= r;

        if (wave == 0) {  // mask-row readout, straight from global (L2-hot)
            float mc = 0.0f;
            #pragma unroll
            for (int v = 0; v < 8; ++v) {
                mc += mrow_base[cb + v + h * 8]      * p0[v];
                mc += mrow_base[cb + 16 + v + h * 8] * p1[v];
            }
            maskacc = maskacc * r + mc;
        }

        // ---- P: C-layout f32 -> B-layout bf16 via per-wave LDS ------------
        __bf16* pbuf = ldsPbase + wave * (QT * CK);   // [n][bank]
        #pragma unroll
        for (int v = 0; v < 8; ++v) {
            pbuf[q * CK +      (v + h * 8)] = (__bf16)p0[v];
            pbuf[q * CK + 16 + (v + h * 8)] = (__bf16)p1[v];
        }
        v16bf Pb;   // same-wave LDS RAW; compiler inserts s_wait_dscnt
        #pragma unroll
        for (int v = 0; v < 8; ++v) {
            const int k = kpair_base(v, h);
            Pb[2 * v]     = pbuf[q * CK + k];
            Pb[2 * v + 1] = pbuf[q * CK + k + 1];
        }

        // ---- PV: each wave accumulates its 128 Dv rows from LDS -----------
        #pragma unroll
        for (int t = 0; t < 8; ++t) {
            v16bf Va;
            const float* vrow = Vbuf + (size_t)(wave * 128 + t * 16 + q) * ROWP;
            #pragma unroll
            for (int v = 0; v < 8; ++v) {
                const float2 f = *(const float2*)&vrow[kpair_base(v, h)];
                Va[2 * v]     = (__bf16)f.x;
                Va[2 * v + 1] = (__bf16)f.y;
            }
            acc[t] = __builtin_amdgcn_wmma_f32_16x16x32_bf16(
                         false, Va, false, Pb, (short)0, acc[t], false, false);
        }

#if USE_TDM
        if (wave == 0 && i + 1 < NCH)
            __builtin_amdgcn_s_wait_tensorcnt((short)0);
#endif
        __syncthreads();   // next buffer ready; current buffer reusable
    }

    // ---- Epilogue: normalize + store --------------------------------------
    const float inv_l = 1.0f / l_run;
    const size_t obase = (size_t)o * (2 * D_VAL) * N_Q;   // out: (1, O, 1024, N)

    #pragma unroll
    for (int t = 0; t < 8; ++t) {
        const int rowb = wave * 128 + t * 16 + h * 8;
        #pragma unroll
        for (int v = 0; v < 8; ++v) {
            out[obase + (size_t)(rowb + v) * N_Q + nbase + q] = acc[t][v] * inv_l;
        }
    }

    if (wave == 0) {
        const float mtot = maskacc + __shfl_xor(maskacc, 16, 32);
        if (h == 0) ldsMaskOut[q] = mtot * inv_l;
    }
    __syncthreads();

    // Gated q_out half: 512 rows x 16 cols over 128 threads.
    for (int i = tid; i < D_VAL * QT; i += NTHREADS) {
        const int row = i >> 4;
        const int col = i & 15;
        const float g  = ldsMaskOut[col];
        const float qv = q_out[(size_t)row * N_Q + nbase + col];
        out[obase + (size_t)(D_VAL + row) * N_Q + nbase + col] = qv * g;
    }
}

extern "C" void kernel_launch(void* const* d_in, const int* in_sizes, int n_in,
                              void* d_out, int out_size, void* d_ws, size_t ws_size,
                              hipStream_t stream) {
    (void)in_sizes; (void)n_in; (void)out_size; (void)d_ws; (void)ws_size;
    const float* keys   = (const float*)d_in[0];
    const float* values = (const float*)d_in[1];
    const float* masks  = (const float*)d_in[2];
    const float* q_in   = (const float*)d_in[3];
    const float* q_out  = (const float*)d_in[4];
    float* out = (float*)d_out;

    dim3 grid(N_Q / QT, OBJ_N);   // 225 x 4 workgroups
    dim3 block(NTHREADS);         // 4 waves (wave32)
    Matcher_48232482734540_kernel<<<grid, block, SMEM_BYTES, stream>>>(
        keys, values, masks, q_in, q_out, out);
}
